// HybridAttention_46669114638425
// MI455X (gfx1250) — matrix operand order repaired and checked by reading, not compiled, not run on hardware
//
#include <hip/hip_runtime.h>

typedef _Float16 v16h __attribute__((ext_vector_type(16)));
typedef _Float16 v8h  __attribute__((ext_vector_type(8)));
typedef _Float16 v4h  __attribute__((ext_vector_type(4)));
typedef float    v8f  __attribute__((ext_vector_type(8)));

typedef unsigned int u32x4 __attribute__((ext_vector_type(4)));
typedef int          i32x4 __attribute__((ext_vector_type(4)));
typedef int          i32x8 __attribute__((ext_vector_type(8)));

#define SEQk      2048
#define NHk       16
#define HDk       64
#define HIDk      1024
#define ROWSk     4096     /* B*S */
#define WIN_HALFk 128

// D = A(16x32 f16) * B(32x16 f16) + C(16x16 f32)  -- CDNA5 wave32 WMMA
static __device__ __forceinline__ v8f wmma_f16(v16h a, v16h b, v8f c) {
  return __builtin_amdgcn_wmma_f32_16x16x32_f16(false, a, false, b, (short)0, c,
                                                false, false);
}

// A-matrix fragment (16x32 f16): lane half g (=lane>=16):
// element e holds K = g*16 + e (contiguous, matches B fragment)
static __device__ __forceinline__ v16h load_a_frag(const _Float16* base, int g) {
  return *(const v16h*)(base + g * 16);
}

// B-matrix fragment (32x16 f16): element e holds K = g*16 + e, N = lane&15.
static __device__ __forceinline__ v16h load_b_frag(const _Float16* base, int g) {
  return *(const v16h*)(base + g * 16);
}

// Issue a 2-D TDM tile copy: global (row-major, 2B elems, row stride = strideElems)
// -> LDS at ldsByteOff.  rowsRemaining = tensor rows from tile start (for OOB
// zero-fill).  Must be executed by a single wave; completion via TENSORcnt.
static __device__ __forceinline__ void tdm_load_2d(unsigned ldsByteOff,
                                                   unsigned long long gaddr,
                                                   unsigned tileRows,
                                                   unsigned rowsRemaining,
                                                   unsigned widthElems,
                                                   unsigned strideElems) {
  u32x4 g0;
  g0[0] = 1u;                                            // count=1, user mode
  g0[1] = ldsByteOff;                                    // lds_addr
  g0[2] = (unsigned)(gaddr & 0xFFFFFFFFull);             // global_addr[31:0]
  g0[3] = (unsigned)((gaddr >> 32) & 0x01FFFFFFull) | (2u << 30);  // [56:32]|type=2
  i32x8 g1;
  g1[0] = (int)(1u << 16);                               // data_size=1 (2 bytes)
  g1[1] = (int)((widthElems & 0xFFFFu) << 16);           // tensor_dim0 lo16
  g1[2] = (int)(((widthElems >> 16) & 0xFFFFu) |
                ((rowsRemaining & 0xFFFFu) << 16));      // t_dim0 hi | t_dim1 lo
  g1[3] = (int)(((rowsRemaining >> 16) & 0xFFFFu) |
                ((widthElems & 0xFFFFu) << 16));         // t_dim1 hi | tile_dim0
  g1[4] = (int)(tileRows & 0xFFFFu);                     // tile_dim1 (tile_dim2=0)
  g1[5] = (int)strideElems;                              // tensor_dim0_stride lo32
  g1[6] = 0;                                             // stride hi | dim1_stride lo
  g1[7] = 0;
  i32x4 g2; g2[0] = g2[1] = g2[2] = g2[3] = 0;           // 2-D: groups 2/3 zero
  i32x4 g3; g3[0] = g3[1] = g3[2] = g3[3] = 0;
#if defined(__clang_major__) && __clang_major__ >= 23
  i32x8 g4; 
#pragma unroll
  for (int e = 0; e < 8; ++e) g4[e] = 0;
  __builtin_amdgcn_tensor_load_to_lds(g0, g1, g2, g3, g4, 0);
#else
  __builtin_amdgcn_tensor_load_to_lds(g0, g1, g2, g3, 0);
#endif
}

// -------------------------------------------------------------------------
// Tiled GEMM: Y[4096x1024] = A[4096x1024] @ W[1024x1024] + bias
// Block tile 128x128, K-step 32, 256 threads (8 waves, 4x2), f16 WMMA, f32 acc.
// -------------------------------------------------------------------------
template <typename AT, bool STORE_HEADS>
__global__ __launch_bounds__(256, 1) void gemm128(const AT* __restrict__ A,
                                                  const float* __restrict__ W,
                                                  const float* __restrict__ bias,
                                                  void* __restrict__ outp) {
  __shared__ __align__(32) _Float16 sA[128 * 32];   // [m][k]
  __shared__ __align__(32) _Float16 sB[128 * 32];   // [n][k] (transposed)

  const int tid  = threadIdx.x;
  const int lane = tid & 31;
  const int wid  = tid >> 5;
  const int g    = lane >> 4;
  const int ln   = lane & 15;
  const int wm   = wid & 3;
  const int wn   = wid >> 2;
  const int bm   = blockIdx.x * 128;
  const int bn   = blockIdx.y * 128;

  v8f acc[2][4];
#pragma unroll
  for (int i = 0; i < 2; ++i)
#pragma unroll
    for (int j = 0; j < 4; ++j)
#pragma unroll
      for (int e = 0; e < 8; ++e) acc[i][j][e] = 0.f;

  for (int k0 = 0; k0 < HIDk; k0 += 32) {
    __syncthreads();
    // ---- stage A tile (128 x 32), f32->f16 (or f16 copy) ----
#pragma unroll
    for (int it = 0; it < 4; ++it) {
      int idx4 = tid + it * 256;
      int row  = idx4 >> 3;
      int kk   = (idx4 & 7) << 2;
      if constexpr (sizeof(AT) == 4) {
        float4 v = *(const float4*)((const float*)A + (size_t)(bm + row) * HIDk + k0 + kk);
        sA[row * 32 + kk + 0] = (_Float16)v.x;
        sA[row * 32 + kk + 1] = (_Float16)v.y;
        sA[row * 32 + kk + 2] = (_Float16)v.z;
        sA[row * 32 + kk + 3] = (_Float16)v.w;
      } else {
        v4h v = *(const v4h*)((const _Float16*)A + (size_t)(bm + row) * HIDk + k0 + kk);
        *(v4h*)&sA[row * 32 + kk] = v;
      }
      if (k0 + 32 < HIDk)
        __builtin_prefetch((const char*)A + (size_t)(bm + row) * HIDk * sizeof(AT) +
                               (size_t)(k0 + 32 + kk) * sizeof(AT), 0, 1);
    }
    // ---- stage B tile transposed: sB[n][k] ----
#pragma unroll
    for (int it = 0; it < 4; ++it) {
      int idx4 = tid + it * 256;
      int kr   = idx4 >> 5;
      int nc   = (idx4 & 31) << 2;
      float4 w = *(const float4*)(W + (size_t)(k0 + kr) * HIDk + bn + nc);
      sB[(nc + 0) * 32 + kr] = (_Float16)w.x;
      sB[(nc + 1) * 32 + kr] = (_Float16)w.y;
      sB[(nc + 2) * 32 + kr] = (_Float16)w.z;
      sB[(nc + 3) * 32 + kr] = (_Float16)w.w;
      if (k0 + 32 < HIDk)
        __builtin_prefetch(W + (size_t)(k0 + 32 + kr) * HIDk + bn + nc, 0, 1);
    }
    __syncthreads();

    v16h af[2], bf[4];
#pragma unroll
    for (int ms = 0; ms < 2; ++ms)
      af[ms] = load_a_frag(&sA[(wm * 32 + ms * 16 + ln) * 32], g);
#pragma unroll
    for (int ns = 0; ns < 4; ++ns)
      bf[ns] = load_b_frag(&sB[(wn * 64 + ns * 16 + ln) * 32], g);
#pragma unroll
    for (int ms = 0; ms < 2; ++ms)
#pragma unroll
      for (int ns = 0; ns < 4; ++ns)
        acc[ms][ns] = wmma_f16(af[ms], bf[ns], acc[ms][ns]);
  }

#pragma unroll
  for (int ms = 0; ms < 2; ++ms) {
#pragma unroll
    for (int ns = 0; ns < 4; ++ns) {
#pragma unroll
      for (int r = 0; r < 8; ++r) {
        int row = bm + wm * 32 + ms * 16 + r + 8 * g;
        int col = bn + wn * 64 + ns * 16 + ln;
        float v = acc[ms][ns][r] + bias[col];
        if constexpr (STORE_HEADS) {
          _Float16* O = (_Float16*)outp;
          int b = row >> 11, s = row & (SEQk - 1);
          int h = col >> 6, d = col & (HDk - 1);
          O[(size_t)(((b * NHk + h) << 17) + (s << 6) + d)] = (_Float16)v;
        } else {
          float* O = (float*)outp;
          O[(size_t)row * HIDk + col] = v;
        }
      }
    }
  }
}

// -------------------------------------------------------------------------
// Sliding-window flash attention. One block = 4 waves = 64 query rows of one
// (b,h); union key span = 320 keys = 10 chunks of 32. K chunk staged by the
// Tensor Data Mover (tensor_load_to_lds + s_wait_tensorcnt); V staged manually
// because the PV B-fragment needs a transposed [d][key] LDS tile.
// -------------------------------------------------------------------------
__global__ __launch_bounds__(128, 1) void attn_swin(const _Float16* __restrict__ Q,
                                                    const _Float16* __restrict__ K,
                                                    const _Float16* __restrict__ V,
                                                    _Float16* __restrict__ AO) {
  __shared__ __align__(32) _Float16 sQ[64 * 64];        // [m][d]
  __shared__ __align__(32) _Float16 sK[32 * 64];        // [key][d]  (TDM dest)
  __shared__ __align__(32) _Float16 sV[64 * 32];        // [d][key] (transposed)
  __shared__ __align__(32) _Float16 sP[4][16 * 32];     // per-wave P tile [m][key]

  const int tid  = threadIdx.x;
  const int lane = tid & 31;
  const int wid  = tid >> 5;
  const int g    = lane >> 4;
  const int ln   = lane & 15;
  const int tile = blockIdx.x & 31;
  const int bh   = blockIdx.x >> 5;
  const int i0   = tile * 64;
  const size_t base = (size_t)bh * SEQk * HDk;

  // stage Q tile (64 x 64 f16): 32 halves (64B) per thread
  {
    int m  = tid >> 1;
    int hp = (tid & 1) * 32;
    const uint4* src = (const uint4*)(Q + base + (size_t)(i0 + m) * HDk + hp);
    uint4* dst = (uint4*)&sQ[m * 64 + hp];
#pragma unroll
    for (int e = 0; e < 4; ++e) dst[e] = src[e];
  }
  __syncthreads();

  v16h qf[2];
#pragma unroll
  for (int wk = 0; wk < 2; ++wk)
    qf[wk] = load_a_frag(&sQ[(wid * 16 + ln) * 64 + wk * 32], g);

  float mrun[8], lrun[8];
  v8f o[4];
#pragma unroll
  for (int r = 0; r < 8; ++r) { mrun[r] = -1e30f; lrun[r] = 0.f; }
#pragma unroll
  for (int nt = 0; nt < 4; ++nt)
#pragma unroll
    for (int e = 0; e < 8; ++e) o[nt][e] = 0.f;

  const int jstart = i0 - WIN_HALFk;   // may be negative
  const unsigned ldsKbase = (unsigned)(unsigned long long)(size_t)(void*)&sK[0];

  for (int c = 0; c < 10; ++c) {
    __syncthreads();   // previous chunk's LDS consumers done
    const int j0c   = jstart + c * 32;               // chunk start key (uniform)
    const int skipR = (j0c < 0) ? ((-j0c < 32) ? -j0c : 32) : 0;
    const int rows  = 32 - skipR;

    // zero-fill the above-tensor rows TDM cannot express (boundary tiles only)
    if (skipR > 0) {
      for (int idx = tid * 4; idx < skipR * 64; idx += 512)
        *(uint2*)&sK[idx] = make_uint2(0u, 0u);
    }
    // ---- stage V chunk manually, transposed [d][key]; zero OOB keys ----
    {
      int t  = tid >> 2;
      int qd = (tid & 3) * 16;
      int j  = j0c + t;
      union { uint4 u[2]; _Float16 h[16]; } vb;
      if (j >= 0 && j < SEQk) {
        const uint4* vs = (const uint4*)(V + base + (size_t)j * HDk + qd);
        vb.u[0] = vs[0]; vb.u[1] = vs[1];
      } else {
        vb.u[0] = make_uint4(0, 0, 0, 0); vb.u[1] = make_uint4(0, 0, 0, 0);
      }
#pragma unroll
      for (int e = 0; e < 16; ++e) sV[(qd + e) * 32 + t] = vb.h[e];
    }
    // ---- stage K chunk via TDM (wave 0 issues; tensor OOB rows auto-zero) ----
    if (wid == 0) {
      if (rows > 0) {
        const int jlo = j0c + skipR;
        tdm_load_2d(ldsKbase + (unsigned)skipR * (HDk * 2),
                    (unsigned long long)(size_t)(K + base + (size_t)jlo * HDk),
                    (unsigned)rows, (unsigned)(SEQk - jlo),
                    (unsigned)HDk, (unsigned)HDk);
      }
      __builtin_amdgcn_s_wait_tensorcnt(0);
    }
    __syncthreads();

    // ---- scores: S = Q * K^T over d=64 (2 K-splits), two 16-key subtiles ----
    v8f s0, s1;
#pragma unroll
    for (int e = 0; e < 8; ++e) { s0[e] = 0.f; s1[e] = 0.f; }
#pragma unroll
    for (int wk = 0; wk < 2; ++wk) {
      v16h k0f = load_b_frag(&sK[ln * 64 + wk * 32], g);
      v16h k1f = load_b_frag(&sK[(16 + ln) * 64 + wk * 32], g);
      s0 = wmma_f16(qf[wk], k0f, s0);
      s1 = wmma_f16(qf[wk], k1f, s1);
    }

    // ---- online softmax (C layout: element r = row r+8g, col = ln) ----
    const int j0 = j0c + ln;
    const int j1 = j0 + 16;
#pragma unroll
    for (int r = 0; r < 8; ++r) {
      const int row = i0 + wid * 16 + r + 8 * g;
      float x0 = s0[r] * 0.125f;   // 1/sqrt(64)
      float x1 = s1[r] * 0.125f;
      if (j0 < row - WIN_HALFk || j0 > row + WIN_HALFk || j0 < 0 || j0 >= SEQk) x0 = -1e30f;
      if (j1 < row - WIN_HALFk || j1 > row + WIN_HALFk || j1 < 0 || j1 >= SEQk) x1 = -1e30f;
      float rm = fmaxf(x0, x1);
#pragma unroll
      for (int off = 1; off < 16; off <<= 1) rm = fmaxf(rm, __shfl_xor(rm, off, 32));
      float mn    = fmaxf(mrun[r], rm);
      float alpha = __expf(mrun[r] - mn);
      float p0    = __expf(x0 - mn);
      float p1    = __expf(x1 - mn);
      float rs    = p0 + p1;
#pragma unroll
      for (int off = 1; off < 16; off <<= 1) rs += __shfl_xor(rs, off, 32);
      lrun[r] = lrun[r] * alpha + rs;
      mrun[r] = mn;
#pragma unroll
      for (int nt = 0; nt < 4; ++nt) o[nt][r] *= alpha;
      sP[wid][(r + 8 * g) * 32 + ln]      = (_Float16)p0;
      sP[wid][(r + 8 * g) * 32 + 16 + ln] = (_Float16)p1;
    }
    // same-wave LDS RAW: LDS is in-order per wave; drain DScnt before re-reading
    asm volatile("s_wait_dscnt 0" ::: "memory");

    // ---- O += P(16x32) * V(32x64), 4 N-subtiles ----
    v16h pf = load_a_frag(&sP[wid][ln * 32], g);
#pragma unroll
    for (int nt = 0; nt < 4; ++nt) {
      v16h vf = load_b_frag(&sV[(nt * 16 + ln) * 32], g);
      o[nt] = wmma_f16(pf, vf, o[nt]);
    }
  }

  // ---- epilogue: normalize and scatter to [B,S,H*D] f16 ----
  const int b = bh >> 4, h = bh & 15;
#pragma unroll
  for (int r = 0; r < 8; ++r) {
    float inv = 1.f / lrun[r];
    int row = i0 + wid * 16 + r + 8 * g;
#pragma unroll
    for (int nt = 0; nt < 4; ++nt) {
      int d = nt * 16 + ln;
      AO[(size_t)(b * SEQk + row) * HIDk + h * HDk + d] = (_Float16)(o[nt][r] * inv);
    }
  }
}

// -------------------------------------------------------------------------
extern "C" void kernel_launch(void* const* d_in, const int* in_sizes, int n_in,
                              void* d_out, int out_size, void* d_ws, size_t ws_size,
                              hipStream_t stream) {
  (void)in_sizes; (void)n_in; (void)out_size; (void)ws_size;
  const float* x  = (const float*)d_in[0];
  const float* wq = (const float*)d_in[1];
  const float* bq = (const float*)d_in[2];
  const float* wk = (const float*)d_in[3];
  const float* bk = (const float*)d_in[4];
  const float* wv = (const float*)d_in[5];
  const float* bv = (const float*)d_in[6];
  const float* wo = (const float*)d_in[7];
  const float* bo = (const float*)d_in[8];

  const size_t perTensor = (size_t)2 * NHk * SEQk * HDk;   // 4M halves = 8MB
  _Float16* qf = (_Float16*)d_ws;
  _Float16* kf = qf + perTensor;
  _Float16* vf = kf + perTensor;
  _Float16* af = vf + perTensor;

  dim3 gg(ROWSk / 128, HIDk / 128);
  gemm128<float, true><<<gg, 256, 0, stream>>>(x, wq, bq, (void*)qf);
  gemm128<float, true><<<gg, 256, 0, stream>>>(x, wk, bk, (void*)kf);
  gemm128<float, true><<<gg, 256, 0, stream>>>(x, wv, bv, (void*)vf);

  attn_swin<<<dim3(2 * NHk * (SEQk / 64)), 128, 0, stream>>>(qf, kf, vf, af);

  gemm128<_Float16, false><<<gg, 256, 0, stream>>>(af, wo, bo, d_out);
}
